// GraphConstruct_spatial_gai_74285754351632
// MI455X (gfx1250) — matrix operand order repaired and checked by reading, not compiled, not run on hardware
//
#include <hip/hip_runtime.h>
#include <hip/hip_bf16.h>
#include <math.h>

// MI455X / gfx1250, wave32. D = ||xe_i - ye_j||, fused bf16-WMMA GEMM + top-16,
// then bandwidth-bound dp construction (512 MB writes -> ~22us roofline floor).

typedef __attribute__((ext_vector_type(16))) __bf16 v16bf;
typedef __attribute__((ext_vector_type(8)))  __bf16 v8bf;
typedef __attribute__((ext_vector_type(8)))  float  v8f;

#define N_CELLS 8192
#define CE 512
#define KNN 16
#define ROWS_PER_BLOCK 32
#define WAVES 8
#define NTILES_PER_WAVE 64   // 8192 cols / 16 per tile / 8 waves

// ---------------- kernel 0: f32 -> bf16 convert + row squared norms ----------------
__global__ void __launch_bounds__(128) cvt_norm_kernel(const float* __restrict__ src,
                                                       __bf16* __restrict__ dst,
                                                       float* __restrict__ nrm) {
  const int row = blockIdx.x;
  const float* s = src + (size_t)row * CE;
  __bf16* d = dst + (size_t)row * CE;
  float acc = 0.f;
  for (int e = threadIdx.x; e < CE; e += 128) {
    float v = s[e];
    acc += v * v;
    d[e] = (__bf16)v;
  }
  __shared__ float red[128];
  red[threadIdx.x] = acc;
  __syncthreads();
  for (int off = 64; off > 0; off >>= 1) {
    if (threadIdx.x < off) red[threadIdx.x] += red[threadIdx.x + off];
    __syncthreads();
  }
  if (threadIdx.x == 0) nrm[row] = red[0];
}

// ---------------- kernel 1: fused bf16 WMMA GEMM + per-row top-16 ----------------
union SmemU {
  __bf16 A[2][16][32][16];                                        // 32 KB: A frags, WMMA lane layout
  struct { float v[32][WAVES][16]; int c[32][WAVES][16]; } mrg;   // 32 KB: merge scratch (reuse)
};

__global__ void __launch_bounds__(256) gemm_topk_kernel(
    const __bf16* __restrict__ xeb, const __bf16* __restrict__ yeb,
    const float* __restrict__ x2g, const float* __restrict__ y2g,
    float* __restrict__ sk_out, float* __restrict__ idxf_out,
    int* __restrict__ idx_ws) {
  __shared__ __align__(32) SmemU smem;
  __shared__ float scr[WAVES][32][16];   // 16 KB: per-wave D^2 tile transpose scratch
  __shared__ float x2s[32];

  const int tid = threadIdx.x;
  const int w = tid >> 5;
  const int lane = tid & 31;
  const int rowBase = blockIdx.x * ROWS_PER_BLOCK;

  // Stage A fragments for 2 M-tiles x 16 k-steps in the lane layout B also uses:
  // lane L holds M = L&15; K = 16*(L>>4)+0..15 (contiguous).
  for (int e = tid; e < 2 * 16 * 32; e += 256) {
    const int mt = e >> 9;
    const int ks = (e >> 5) & 15;
    const int ln = e & 31;
    const int srow = rowBase + mt * 16 + (ln & 15);
    const int kb0 = ks * 32 + 16 * (ln >> 4);
    const __bf16* s = xeb + (size_t)srow * CE;
    __bf16* d = &smem.A[mt][ks][ln][0];
    *(v8bf*)(d)     = *(const v8bf*)(s + kb0);
    *(v8bf*)(d + 8) = *(const v8bf*)(s + kb0 + 8);
  }
  if (tid < 32) x2s[tid] = x2g[rowBase + tid];
  __syncthreads();

  const int half = lane >> 4;
  const int nc = lane & 15;
  const int kb = 16 * half;   // B frag: lane L holds N=L&15, K = 16*(L>>4)+0..15 (contiguous)

  // Per-lane running top-16 (lane L owns row L of the 32-row block),
  // kept sorted descending in registers (all indices compile-time -> stays in VGPRs).
  float tv[16]; int tc[16];
  #pragma unroll
  for (int i = 0; i < 16; ++i) { tv[i] = 3.0e38f; tc[i] = 0; }

  for (int t = 0; t < NTILES_PER_WAVE; ++t) {
    const int n0 = (t * WAVES + w) * 16;       // interleaved column tiles across waves
    const int nrow = n0 + nc;                  // ye row == B column
    const __bf16* bp = yeb + (size_t)nrow * CE + kb;
    const float y2c = y2g[nrow];

    v8f acc0 = {}; v8f acc1 = {};
    #pragma unroll
    for (int ks = 0; ks < 16; ++ks) {
      v16bf b  = *(const v16bf*)(bp + ks * 32);
      v16bf a0 = *(const v16bf*)(&smem.A[0][ks][lane][0]);
      v16bf a1 = *(const v16bf*)(&smem.A[1][ks][lane][0]);
      acc0 = __builtin_amdgcn_wmma_f32_16x16x32_bf16(false, a0, false, b,
                                                     (short)0, acc0, false, false);
      acc1 = __builtin_amdgcn_wmma_f32_16x16x32_bf16(false, a1, false, b,
                                                     (short)0, acc1, false, false);
    }

    // C layout: lane L holds N=L&15, VGPR v holds M = v + 8*(L>>4).
    // Transpose D^2 tile through LDS so lane L gets row L's 16 candidates.
    #pragma unroll
    for (int v = 0; v < 8; ++v) {
      const int r0 = v + 8 * half;
      scr[w][r0][nc]      = x2s[r0]      + y2c - 2.f * acc0[v];
      scr[w][16 + r0][nc] = x2s[16 + r0] + y2c - 2.f * acc1[v];
    }
    asm volatile("s_wait_dscnt 0x0" ::: "memory");  // same-wave LDS RAW fence

    #pragma unroll
    for (int j = 0; j < 16; ++j) {
      const float d2 = scr[w][lane][j];
      if (d2 < tv[0]) {                 // guarded insert: common path is 1 compare
        tv[0] = d2; tc[0] = n0 + j;
        #pragma unroll
        for (int i = 0; i < 15; ++i) {
          if (tv[i] < tv[i + 1]) {
            float ft = tv[i]; tv[i] = tv[i + 1]; tv[i + 1] = ft;
            int it = tc[i]; tc[i] = tc[i + 1]; tc[i + 1] = it;
          }
        }
      }
    }
  }

  __syncthreads();   // all waves done with smem.A -> safe to reuse for merge
  #pragma unroll
  for (int s = 0; s < 16; ++s) {
    smem.mrg.v[lane][w][s] = tv[s];
    smem.mrg.c[lane][w][s] = tc[s];
  }
  __syncthreads();

  // Final 16-of-128 ascending selection, one lane per row (wave 0).
  if (tid < 32) {
    const int grow = rowBase + tid;
    float* mv = &smem.mrg.v[tid][0][0];
    int*   mc = &smem.mrg.c[tid][0][0];
    for (int p = 0; p < KNN; ++p) {
      float best = mv[0]; int bi = 0;
      for (int j = 1; j < WAVES * 16; ++j) {
        float v = mv[j];
        if (v < best) { best = v; bi = j; }
      }
      const int bcol = mc[bi];
      mv[bi] = 3.4e38f;
      const float dist = sqrtf(fmaxf(best, 0.f));
      const float skv = __expf(dist * -0.1f);
      // sk: (1, 16, 2*8192): [p][ss*8192 + row]
      sk_out[p * (2 * N_CELLS) + grow] = skv;
      sk_out[p * (2 * N_CELLS) + N_CELLS + grow] = skv;
      // idx_k: (1, 8192, 16) emitted as float values
      idxf_out[(size_t)grow * KNN + p] = (float)bcol;
      idx_ws[grow * KNN + p] = bcol;
    }
  }
}

// ---------------- kernel 2: dp = |ye[mm,ee] - xe[idx[mm,kk],ee]|, 512 MB streamed ----------------
__global__ void __launch_bounds__(256) dp_kernel(const float* __restrict__ xe,
                                                 const float* __restrict__ ye,
                                                 const int* __restrict__ idx,
                                                 float* __restrict__ dp) {
  const int mm = blockIdx.x * 256 + threadIdx.x;   // fast dim -> coalesced stores
  const int ee = blockIdx.y;
  const int kk = blockIdx.z;
  const int id = idx[mm * KNN + kk];
  const float v = fabsf(ye[(size_t)mm * CE + ee] - xe[(size_t)id * CE + ee]);
  const size_t base = (size_t)(kk * CE + ee) * (2 * N_CELLS);
  dp[base + mm] = v;               // ss = 0
  dp[base + N_CELLS + mm] = v;     // ss = 1 (scale-2 duplicate)
}

extern "C" void kernel_launch(void* const* d_in, const int* in_sizes, int n_in,
                              void* d_out, int out_size, void* d_ws, size_t ws_size,
                              hipStream_t stream) {
  const float* xe = (const float*)d_in[0];
  const float* ye = (const float*)d_in[1];
  // d_in[2] (spatial) is unused by the reference forward.
  float* out = (float*)d_out;

  // Workspace layout (~17 MB):
  char* ws = (char*)d_ws;
  __bf16* xeb = (__bf16*)(ws);                       // 8 MB
  __bf16* yeb = (__bf16*)(ws + 8388608);             // 8 MB
  float*  x2  = (float*)(ws + 16777216);             // 32 KB
  float*  y2  = (float*)(ws + 16809984);             // 32 KB
  int*    idxw = (int*)(ws + 16842752);              // 512 KB

  // Output layout (concatenated, return order): sk | idx_k | dp
  float* sk_out   = out;                 // 16 * 16384       = 262144
  float* idxf_out = out + 262144;        // 8192 * 16        = 131072
  float* dp_out   = out + 393216;        // 16*512*16384     = 134217728

  cvt_norm_kernel<<<N_CELLS, 128, 0, stream>>>(xe, xeb, x2);
  cvt_norm_kernel<<<N_CELLS, 128, 0, stream>>>(ye, yeb, y2);

  gemm_topk_kernel<<<N_CELLS / ROWS_PER_BLOCK, 256, 0, stream>>>(
      xeb, yeb, x2, y2, sk_out, idxf_out, idxw);

  dp_kernel<<<dim3(N_CELLS / 256, CE, KNN), 256, 0, stream>>>(xe, ye, idxw, dp_out);
}
